// SharedGraphModelling_56831007260746
// MI455X (gfx1250) — matrix operand order repaired and checked by reading, not compiled, not run on hardware
//
#include <hip/hip_runtime.h>

#define NNODES 4096
#define NSUBS  1024
#define NFEAT  128
#define HID    512
#define NHEADS 8
#define DHEAD  64

typedef __attribute__((ext_vector_type(16))) _Float16 v16h;
typedef __attribute__((ext_vector_type(8)))  _Float16 v8h;
typedef __attribute__((ext_vector_type(8)))  float    v8f;

// f16 WMMA operand ownership (CDNA5 ISA 7.12.2), used throughout:
//   A (16x32): lanes 0-15 (M=lane)    own K = {0..7, 16..23}
//              lanes 16-31 (M=lane-16) own K = {8..15, 24..31}
//   => per lane: two contiguous 8-element runs at K = hi and K = 16+hi,
//      hi = (lane&16) ? 8 : 0.
//   B (32x16): lane = N column (lane&15); same K runs as A:
//      K = hi..hi+7 and 16+hi..16+hi+7.

// ---------------------------------------------------------------------------
// Tiled mixed-precision GEMM: C[M,N] = A[M,K] @ B[K,N] (+bias), f32 in/out,
// f16 operands into v_wmma_f32_16x16x32_f16 with f32 accumulation.
// Tiles staged in LDS pre-converted to f16 so each WMMA operand is built from
// two aligned 16B LDS vector loads. Optional fused epilogue writes a
// transposed f16 copy CT16[n*M+m] (K-contiguous B operands for attention).
// Block = 256 threads = 8 waves; macro-tile 32(M) x 64(N); wave tile 16x16.
// ---------------------------------------------------------------------------
__global__ void gemm_wmma(const float* __restrict__ A, const float* __restrict__ B,
                          const float* __restrict__ bias, float* __restrict__ C,
                          _Float16* __restrict__ CT16, int M, int N, int K) {
  __shared__ __align__(32) _Float16 As16[32][40];   // [m][k], row stride 80B
  __shared__ __align__(32) _Float16 Bs16[64][40];   // [n][k] (transposed)
  const int tid  = threadIdx.x;
  const int lane = tid & 31;
  const int w    = tid >> 5;       // 0..7
  const int wm   = w >> 2;         // 0..1
  const int wn   = w & 3;          // 0..3
  const int m0   = blockIdx.x * 32;
  const int n0   = blockIdx.y * 64;
  const int ar   = wm * 16 + (lane & 15);
  const int bc   = wn * 16 + (lane & 15);
  const int hi   = (lane & 16) ? 8 : 0;
  const int bk   = (lane & 16) ? 16 : 0;

  v8f acc = {};
  for (int kb = 0; kb < K; kb += 32) {
    // stage A 32x32 (f32 -> f16), pairwise so stores merge
    for (int p = tid; p < 512; p += 256) {
      int r = p >> 4, c2 = (p & 15) * 2;
      As16[r][c2]     = (_Float16)A[(size_t)(m0 + r) * K + kb + c2];
      As16[r][c2 + 1] = (_Float16)A[(size_t)(m0 + r) * K + kb + c2 + 1];
    }
    // stage B 32(k) x 64(n) transposed -> [n][k]
    for (int i = tid; i < 2048; i += 256) {
      int r = i >> 6, c = i & 63;
      Bs16[c][r] = (_Float16)B[(size_t)(kb + r) * N + n0 + c];
    }
    __syncthreads();

    v8h alo = *(const v8h*)&As16[ar][hi];
    v8h ahi = *(const v8h*)&As16[ar][16 + hi];
    v16h av = (v16h)__builtin_shufflevector(alo, ahi, 0, 1, 2, 3, 4, 5, 6, 7,
                                            8, 9, 10, 11, 12, 13, 14, 15);
    v8h blo = *(const v8h*)&Bs16[bc][hi];
    v8h bhi = *(const v8h*)&Bs16[bc][16 + hi];
    v16h bv = (v16h)__builtin_shufflevector(blo, bhi, 0, 1, 2, 3, 4, 5, 6, 7,
                                            8, 9, 10, 11, 12, 13, 14, 15);

    acc = __builtin_amdgcn_wmma_f32_16x16x32_f16(false, av, false, bv,
                                                 (short)0, acc, false, false);
    __syncthreads();
  }

  const int col = n0 + wn * 16 + (lane & 15);
  const float bc0 = bias ? bias[col] : 0.0f;
#pragma unroll
  for (int v = 0; v < 8; ++v) {
    int row = m0 + wm * 16 + v + ((lane & 16) ? 8 : 0);
    float o = acc[v] + bc0;
    C[(size_t)row * N + col] = o;
    if (CT16) CT16[(size_t)col * M + row] = (_Float16)o;
  }
}

// ---------------------------------------------------------------------------
// f_src[h,n] = dot(Wh[n, h*64:h*64+64], a_src[h]);  f_dst likewise.
// ---------------------------------------------------------------------------
__global__ void attn_coeff(const float* __restrict__ Wh, const float* __restrict__ asrc,
                           const float* __restrict__ adst, float* __restrict__ fsrc,
                           float* __restrict__ fdst, int N) {
  int idx = blockIdx.x * blockDim.x + threadIdx.x;   // idx = h*N + n
  if (idx >= NHEADS * N) return;
  int h = idx / N, n = idx - h * N;
  const float* wrow = Wh + (size_t)n * HID + h * DHEAD;
  float s0 = 0.f, s1 = 0.f;
#pragma unroll 4
  for (int d = 0; d < DHEAD; ++d) {
    float v = wrow[d];
    s0 += v * asrc[h * DHEAD + d];
    s1 += v * adst[h * DHEAD + d];
  }
  fsrc[idx] = s0;
  fdst[idx] = s1;
}

// ---------------------------------------------------------------------------
// Fused masked-GAT attention, flash style (never materializes e[H,N,N]).
// Block = 256 threads = 8 waves, wave w == head w; block handles 16 rows.
// Adjacency tiles are staged LDS-direct with GLOBAL_LOAD_ASYNC_TO_LDS_B64
// (ASYNCcnt-tracked, no VGPR round trip), shared by all 8 heads. Scores +
// online softmax in VALU/shuffles; P(f16) @ Wh(f16) via 4 WMMAs into f32
// accumulators. Epilogue: /l then ELU.
// ---------------------------------------------------------------------------
__global__ void gat_attn(const float* __restrict__ adj, const float* __restrict__ fsrc,
                         const float* __restrict__ fdst, const _Float16* __restrict__ WhT,
                         float* __restrict__ out, int N) {
  __shared__ __align__(32) float adjS[16][40];      // 16x32 tile, row stride 160B
  __shared__ __align__(32) float fdS[NHEADS][32];

  const int tid  = threadIdx.x;
  const int lane = tid & 31;
  const int h    = tid >> 5;            // head
  const int r    = lane & 15;           // row within tile owned by this lane
  const int hi   = (lane & 16) ? 8 : 0; // K-run base (f16 operand layout)
  const int i0   = blockIdx.x * 16;

  // async-copy slot for this thread: 2 consecutive floats of the 16x32 tile
  const int arow = tid >> 4;            // 0..15
  const int acol = (tid & 15) * 2;      // 0,2,..,30
  const unsigned ldsa = (unsigned)(uintptr_t)&adjS[arow][acol];
  const unsigned long long abase =
      (unsigned long long)(uintptr_t)(adj + (size_t)i0 * N);

  const float fsr = fsrc[(size_t)h * N + i0 + r];
  float m = -1e30f, l = 0.0f;
  v8f acc[4] = {};

  for (int jt = 0; jt < N; jt += 32) {
    __syncthreads();
    // adjacency tile: memory -> LDS without touching VGPRs (ASYNCcnt)
    unsigned voff = (unsigned)(((size_t)arow * N + jt + acol) * sizeof(float));
    asm volatile("global_load_async_to_lds_b64 %0, %1, %2"
                 :: "v"(ldsa), "v"(voff), "s"(abase) : "memory");
    // f_dst slice for all heads for this j-tile
    fdS[h][lane] = fdst[(size_t)h * N + jt + lane];
    // prefetch next Wh panel (global_prefetch_b8)
    if (jt + 32 < N)
      __builtin_prefetch(WhT + (size_t)(h * DHEAD + r) * N + jt + 32, 0, 1);
    asm volatile("s_wait_asynccnt 0x0" ::: "memory");
    __syncthreads();

    // scores for the 16 operand elements this lane owns: two contiguous runs
    v8f alo = *(const v8f*)&adjS[r][hi];
    v8f ahi = *(const v8f*)&adjS[r][16 + hi];
    v8f flo = *(const v8f*)&fdS[h][hi];
    v8f fhi = *(const v8f*)&fdS[h][16 + hi];
    float sc[16];
    float tmax = -1e30f;
#pragma unroll
    for (int e = 0; e < 8; ++e) {
      float s = fsr + flo[e];
      s = (s > 0.f) ? s : 0.2f * s;                 // leaky_relu(0.2)
      s = (alo[e] > 0.f) ? s : -1e9f;               // mask
      sc[e] = s;
      tmax = fmaxf(tmax, s);
    }
#pragma unroll
    for (int e = 0; e < 8; ++e) {
      float s = fsr + fhi[e];
      s = (s > 0.f) ? s : 0.2f * s;
      s = (ahi[e] > 0.f) ? s : -1e9f;
      sc[8 + e] = s;
      tmax = fmaxf(tmax, s);
    }
    tmax = fmaxf(tmax, __shfl_xor(tmax, 16, 32));   // combine lane pair (r, r+16)
    float mnew  = fmaxf(m, tmax);
    float scale = __expf(m - mnew);

    float ps = 0.f;
    v16h pa;
#pragma unroll
    for (int e = 0; e < 16; ++e) {
      float p = __expf(sc[e] - mnew);
      ps += p;
      pa[e] = (_Float16)p;
    }
    ps += __shfl_xor(ps, 16, 32);
    l = l * scale + ps;
    m = mnew;

    // rescale accumulators by per-row scale (row q's state lives in lane q)
    const int hi8 = (lane & 16) ? 8 : 0;
#pragma unroll
    for (int v = 0; v < 8; ++v) {
      float sv = __shfl(scale, v + hi8, 32);
      acc[0][v] *= sv; acc[1][v] *= sv; acc[2][v] *= sv; acc[3][v] *= sv;
    }

    // P(16x32,f16) @ Wh(32x64,f16) -> 4 WMMAs, f32 accumulate.
    const int bko = (lane & 16) ? 16 : 0;
#pragma unroll
    for (int ct = 0; ct < 4; ++ct) {
      const _Float16* bp = WhT + (size_t)(h * DHEAD + ct * 16 + r) * N + jt + hi;
      v8h b0 = *reinterpret_cast<const v8h*>(bp);
      v8h b1 = *reinterpret_cast<const v8h*>(bp + 16);
      v16h bv = __builtin_shufflevector(b0, b1, 0, 1, 2, 3, 4, 5, 6, 7,
                                        8, 9, 10, 11, 12, 13, 14, 15);
      acc[ct] = __builtin_amdgcn_wmma_f32_16x16x32_f16(false, pa, false, bv,
                                                       (short)0, acc[ct], false, false);
    }
  }

  // epilogue: divide by l, ELU, store
  const int hi8 = (lane & 16) ? 8 : 0;
#pragma unroll
  for (int v = 0; v < 8; ++v) {
    float lv  = __shfl(l, v + hi8, 32);
    float inv = 1.0f / lv;
    int row   = i0 + v + hi8;
#pragma unroll
    for (int ct = 0; ct < 4; ++ct) {
      float o = acc[ct][v] * inv;
      o = (o > 0.f) ? o : (__expf(o) - 1.0f);       // ELU
      out[(size_t)row * HID + h * DHEAD + ct * 16 + r] = o;
    }
  }
}

// ---------------------------------------------------------------------------
// Segment-mean pooling
// ---------------------------------------------------------------------------
__global__ void pool_zero(float* __restrict__ sums, float* __restrict__ cnt) {
  int idx = blockIdx.x * blockDim.x + threadIdx.x;
  if (idx < NSUBS * HID) sums[idx] = 0.f;
  if (idx < NSUBS) cnt[idx] = 0.f;
}

__global__ void pool_accum(const float* __restrict__ hin, const int* __restrict__ seg,
                           float* __restrict__ sums, float* __restrict__ cnt, int N) {
  int idx = blockIdx.x * blockDim.x + threadIdx.x;
  if (idx >= N * HID) return;
  int n = idx >> 9, c = idx & (HID - 1);
  int s = seg[n];
  atomicAdd(&sums[(size_t)s * HID + c], hin[idx]);
  if (c == 0) atomicAdd(&cnt[s], 1.0f);
}

__global__ void pool_div(float* __restrict__ sums, const float* __restrict__ cnt) {
  int idx = blockIdx.x * blockDim.x + threadIdx.x;
  if (idx >= NSUBS * HID) return;
  sums[idx] /= fmaxf(cnt[idx >> 9], 1.0f);
}

// ---------------------------------------------------------------------------
extern "C" void kernel_launch(void* const* d_in, const int* in_sizes, int n_in,
                              void* d_out, int out_size, void* d_ws, size_t ws_size,
                              hipStream_t stream) {
  (void)in_sizes; (void)n_in; (void)out_size; (void)ws_size;
  const float* x         = (const float*)d_in[0];
  const float* adj_node  = (const float*)d_in[1];
  const float* adj_sub   = (const float*)d_in[2];
  const int*   seg_ids   = (const int*)d_in[3];
  const float* lin_w     = (const float*)d_in[4];
  const float* lin_b     = (const float*)d_in[5];
  const float* node_W    = (const float*)d_in[6];
  const float* node_asrc = (const float*)d_in[7];
  const float* node_adst = (const float*)d_in[8];
  const float* sub_W     = (const float*)d_in[9];
  const float* sub_asrc  = (const float*)d_in[10];
  const float* sub_adst  = (const float*)d_in[11];

  // Reference bug: each layer reads the ORIGINAL input; only the last layer's
  // output survives. So only layer index 1 of each stack matters.
  const float* nW1 = node_W    + HID * HID;
  const float* nA1 = node_asrc + NHEADS * DHEAD;
  const float* nD1 = node_adst + NHEADS * DHEAD;
  const float* sW1 = sub_W     + HID * HID;
  const float* sA1 = sub_asrc  + NHEADS * DHEAD;
  const float* sD1 = sub_adst  + NHEADS * DHEAD;

  char* ws = (char*)d_ws;
  size_t off = 0;
  auto alloc = [&](size_t bytes) -> void* {
    void* p = ws + off;
    off = (off + bytes + 255) & ~(size_t)255;
    return p;
  };
  float* hbuf = (float*)alloc((size_t)NNODES * HID * 4);
  float* WhN  = (float*)alloc((size_t)NNODES * HID * 4);
  float* h2   = (float*)alloc((size_t)NNODES * HID * 4);
  float* pool = (float*)alloc((size_t)NSUBS * HID * 4);
  float* WhS  = (float*)alloc((size_t)NSUBS * HID * 4);
  float* cnt  = (float*)alloc((size_t)NSUBS * 4);
  float* fsN  = (float*)alloc((size_t)NHEADS * NNODES * 4);
  float* fdN  = (float*)alloc((size_t)NHEADS * NNODES * 4);
  float* fsS  = (float*)alloc((size_t)NHEADS * NSUBS * 4);
  float* fdSb = (float*)alloc((size_t)NHEADS * NSUBS * 4);
  _Float16* WhTN = (_Float16*)alloc((size_t)HID * NNODES * 2);
  _Float16* WhTS = (_Float16*)alloc((size_t)HID * NSUBS * 2);

  // 1) h = x @ lin_w + b     (4096x128x512)
  gemm_wmma<<<dim3(NNODES / 32, HID / 64), 256, 0, stream>>>(
      x, lin_w, lin_b, hbuf, nullptr, NNODES, HID, NFEAT);

  // 2) Wh = h @ node_W[1]    (4096x512x512) + f16 transposed copy
  gemm_wmma<<<dim3(NNODES / 32, HID / 64), 256, 0, stream>>>(
      hbuf, nW1, nullptr, WhN, WhTN, NNODES, HID, HID);

  // 3) attention coefficients f_src/f_dst
  attn_coeff<<<(NHEADS * NNODES + 255) / 256, 256, 0, stream>>>(
      WhN, nA1, nD1, fsN, fdN, NNODES);

  // 4) fused masked softmax + alpha@Wh + ELU
  gat_attn<<<NNODES / 16, 256, 0, stream>>>(adj_node, fsN, fdN, WhTN, h2, NNODES);

  // 5) segment mean to substations
  pool_zero<<<(NSUBS * HID + 255) / 256, 256, 0, stream>>>(pool, cnt);
  pool_accum<<<(NNODES * HID + 255) / 256, 256, 0, stream>>>(h2, seg_ids, pool, cnt, NNODES);
  pool_div<<<(NSUBS * HID + 255) / 256, 256, 0, stream>>>(pool, cnt);

  // 6) Wh_sub = pool @ sub_W[1]   (1024x512x512)
  gemm_wmma<<<dim3(NSUBS / 32, HID / 64), 256, 0, stream>>>(
      pool, sW1, nullptr, WhS, WhTS, NSUBS, HID, HID);

  // 7) sub attention coefficients
  attn_coeff<<<(NHEADS * NSUBS + 255) / 256, 256, 0, stream>>>(
      WhS, sA1, sD1, fsS, fdSb, NSUBS);

  // 8) fused sub-GAT -> final output [1024, 512]
  gat_attn<<<NSUBS / 16, 256, 0, stream>>>(adj_sub, fsS, fdSb, WhTS, (float*)d_out, NSUBS);
}
